// Digital_update_36867999269163
// MI455X (gfx1250) — matrix operand order repaired and checked by reading, not compiled, not run on hardware
//
#include <hip/hip_runtime.h>
#include <hip/hip_bf16.h>
#include <math.h>

typedef __attribute__((ext_vector_type(16))) _Float16 v16h;
typedef __attribute__((ext_vector_type(8)))  _Float16 v8h;
typedef __attribute__((ext_vector_type(8)))  float    v8f;

#define BATCH 512
#define NUM_M 64
#define NUM_USER 32
#define FEAT_B 62
#define MROWS (BATCH * NUM_USER)   // 16384
#define K0 64                      // 2 + 62
#define NHID 512

#define TILE_M 128
#define TILE_N 128
#define KSTEP  32

// ---------------------------------------------------------------------------
// CDNA5 async global->LDS copy (16B per lane), tracked by ASYNCcnt.
// DMA writes LDS directly: no VGPR round-trip, no s_wait_loadcnt stall.
// ---------------------------------------------------------------------------
__device__ __forceinline__ void async_copy_b128(const void* g, void* lds)
{
    // generic pointer to LDS: low 32 bits are the LDS byte offset
    unsigned lds_off = (unsigned)(unsigned long long)lds;
    asm volatile("global_load_async_to_lds_b128 %0, %1, off"
                 :: "v"(lds_off), "v"(g)
                 : "memory");
}
#define WAIT_ASYNCCNT(n) asm volatile("s_wait_asynccnt " #n ::: "memory")

// ---------------------------------------------------------------------------
// Kernel 1: X[row][64] (f16) = [ D[b,u,0:2] | sum_m B[b,m,u,0:62] ]
// One block per batch. float2 loads (62 = 2*31, rows 8B-aligned), coalesced
// over the feature dim. This is the bandwidth-dominant pass (260 MB of B).
// ---------------------------------------------------------------------------
__global__ __launch_bounds__(256) void build_x_kernel(
    const float* __restrict__ D, const float* __restrict__ B,
    _Float16* __restrict__ X)
{
    const int b   = blockIdx.x;
    const int tid = threadIdx.x;

    // D part: 32 users x 2 features
    if (tid < NUM_USER * 2) {
        const int u = tid >> 1, c = tid & 1;
        X[(size_t)(b * NUM_USER + u) * K0 + c] =
            (_Float16)D[(size_t)(b * NUM_USER + u) * 2 + c];
    }

    // B reduction over antennas, two features at a time
    const float2* B2 = (const float2*)B;
    for (int p = tid; p < NUM_USER * 31; p += 256) {
        const int u = p / 31, f2 = p % 31;
        const float2* bp =
            B2 + ((size_t)b * NUM_M * NUM_USER + u) * 31 + f2;
        float ax = 0.f, ay = 0.f;
        #pragma unroll 8
        for (int m = 0; m < NUM_M; ++m) {
            float2 v = bp[(size_t)m * NUM_USER * 31];
            ax += v.x; ay += v.y;
        }
        _Float16* xp = X + (size_t)(b * NUM_USER + u) * K0 + 2 + f2 * 2;
        xp[0] = (_Float16)ax;
        xp[1] = (_Float16)ay;
    }
}

// ---------------------------------------------------------------------------
// Kernel 2: weight convert + transpose  W[K][N] f32  ->  Wt[N][K] f16
// ---------------------------------------------------------------------------
__global__ __launch_bounds__(256) void convert_wt_kernel(
    const float* __restrict__ W, _Float16* __restrict__ Wt, int K, int N)
{
    const int idx = blockIdx.x * 256 + threadIdx.x;
    if (idx < K * N) {
        const int k = idx / N, n = idx % N;
        Wt[(size_t)n * K + k] = (_Float16)W[idx];
    }
}

// ---------------------------------------------------------------------------
// Kernel 3: WMMA GEMM  Out = relu(X @ W + bias), f16 in, f32 acc, f16 out.
// Block tile 128x128, 8 waves (wave32), per-wave 32x64 = 8 WMMA acc tiles.
// Double-buffered async global->LDS staging; last k-step peeled so the
// steady-state loop body is branch-free.
// ---------------------------------------------------------------------------
__global__ __launch_bounds__(256) void gemm_relu_wmma_kernel(
    const _Float16* __restrict__ Xg,   // [M][K] row-major f16
    const _Float16* __restrict__ Wt,   // [N][K] row-major f16 (W transposed)
    const float*    __restrict__ bias, // [N] f32
    _Float16*       __restrict__ Out,  // [M][N] f16
    int K, int N)
{
    __shared__ __align__(16) _Float16 As[2][TILE_M][KSTEP];
    __shared__ __align__(16) _Float16 Bs[2][TILE_N][KSTEP];

    const int tid   = threadIdx.x;
    const int lane  = tid & 31;
    const int wave  = tid >> 5;
    const int waveM = wave >> 1;   // 0..3 -> 32-row slab
    const int waveN = wave & 1;    // 0..1 -> 64-col slab
    const int m_blk = blockIdx.y * TILE_M;
    const int n_blk = blockIdx.x * TILE_N;

    // Per-stage: 512 16B chunks for A, 512 for B; each thread owns 2 + 2.
    const int r0 = tid >> 2;             // chunk row for j=0
    const int c0 = tid & 3;              // chunk col (8 halves)
    auto stage = [&](int k0, int buf) {
        #pragma unroll
        for (int j = 0; j < 2; ++j) {
            const int r = r0 + j * 64;
            async_copy_b128(&Xg[(size_t)(m_blk + r) * K + k0 + c0 * 8],
                            &As[buf][r][c0 * 8]);
        }
        #pragma unroll
        for (int j = 0; j < 2; ++j) {
            const int r = r0 + j * 64;
            async_copy_b128(&Wt[(size_t)(n_blk + r) * K + k0 + c0 * 8],
                            &Bs[buf][r][c0 * 8]);
        }
    };

    v8f acc[2][4] = {};

    auto compute = [&](int buf) {
        // A fragments: lane<16 -> K 0..7 & 16..23 ; lane>=16 -> 8..15 & 24..31
        v16h afrag[2], bfrag[4];
        const int mrow = lane & 15;
        const int ka   = (lane >> 4) * 8;
        #pragma unroll
        for (int tm = 0; tm < 2; ++tm) {
            const _Float16* ap = &As[buf][waveM * 32 + tm * 16 + mrow][0];
            v8h lo = *(const v8h*)(ap + ka);
            v8h hi = *(const v8h*)(ap + ka + 16);
            afrag[tm] = __builtin_shufflevector(lo, hi,
                0,1,2,3,4,5,6,7,8,9,10,11,12,13,14,15);
        }
        // B fragments: lane<16 -> K 0..7 & 16..23 at N=lane ; lane>=16 -> K 8..15 & 24..31
        const int kb = (lane >> 4) * 8;
        #pragma unroll
        for (int tn = 0; tn < 4; ++tn) {
            const _Float16* bp = &Bs[buf][waveN * 64 + tn * 16 + (lane & 15)][0];
            v8h lo = *(const v8h*)(bp + kb);
            v8h hi = *(const v8h*)(bp + kb + 16);
            bfrag[tn] = __builtin_shufflevector(lo, hi,
                0,1,2,3,4,5,6,7,8,9,10,11,12,13,14,15);
        }
        #pragma unroll
        for (int tm = 0; tm < 2; ++tm)
            #pragma unroll
            for (int tn = 0; tn < 4; ++tn)
                acc[tm][tn] = __builtin_amdgcn_wmma_f32_16x16x32_f16(
                    /*neg_a=*/false, afrag[tm], /*neg_b=*/false, bfrag[tn],
                    /*c_mod=*/(short)0, acc[tm][tn],
                    /*reuse_a=*/false, /*reuse_b=*/false);
    };

    const int nsteps = K / KSTEP;
    stage(0, 0);

    // steady state: branch-free stage -> wait(4) -> barrier -> wmma -> barrier
    for (int s = 0; s < nsteps - 1; ++s) {
        stage((s + 1) * KSTEP, (s + 1) & 1);  // prefetch next tile
        WAIT_ASYNCCNT(0x4);                   // retire current tile's 4 loads
        __syncthreads();
        compute(s & 1);
        __syncthreads();                      // reads done before overwrite
    }
    // peeled last step
    WAIT_ASYNCCNT(0x0);
    __syncthreads();
    compute((nsteps - 1) & 1);

    // ---- epilogue: C/D layout -> VGPR r: lanes 0-15 row r, lanes 16-31 row r+8
    const int rowadd = (lane >> 4) * 8;
    const int ncol   = lane & 15;
    #pragma unroll
    for (int tm = 0; tm < 2; ++tm) {
        #pragma unroll
        for (int tn = 0; tn < 4; ++tn) {
            const int n    = n_blk + waveN * 64 + tn * 16 + ncol;
            const float bv = bias[n];
            const int mbase = m_blk + waveM * 32 + tm * 16 + rowadd;
            #pragma unroll
            for (int r = 0; r < 8; ++r) {
                float v = acc[tm][tn][r] + bv;
                v = v > 0.f ? v : 0.f;   // ReLU
                Out[(size_t)(mbase + r) * N + n] = (_Float16)v;
            }
        }
    }
}

// ---------------------------------------------------------------------------
// Kernel 4: head  D1 = sigmoid(H3 @ W4 + b4), per-batch normalize over users.
// One block per batch (64 threads = 32 users x 2 outputs).
// ---------------------------------------------------------------------------
__global__ __launch_bounds__(64) void head_norm_kernel(
    const _Float16* __restrict__ H3, const float* __restrict__ W4,
    const float* __restrict__ b4, const float* __restrict__ Ppow,
    float* __restrict__ out)
{
    __shared__ float w4s[NHID * 2];
    __shared__ float d1s[NUM_USER * 2];
    __shared__ float ssum[2];

    const int b = blockIdx.x;
    const int tid = threadIdx.x;

    for (int i = tid; i < NHID * 2; i += 64) w4s[i] = W4[i];
    __syncthreads();

    const int u = tid >> 1, o = tid & 1;
    const v8h* h8 = (const v8h*)(H3 + (size_t)(b * NUM_USER + u) * NHID);
    float acc = b4[o];
    for (int k8 = 0; k8 < NHID / 8; ++k8) {
        v8h hv = h8[k8];                       // 16B vector load
        #pragma unroll
        for (int j = 0; j < 8; ++j)
            acc += (float)hv[j] * w4s[(k8 * 8 + j) * 2 + o];
    }
    const float s = 1.f / (1.f + __expf(-acc));
    d1s[u * 2 + o] = s;
    __syncthreads();

    if (tid < 2) {
        float t = 0.f;
        for (int uu = 0; uu < NUM_USER; ++uu) t += d1s[uu * 2 + tid];
        ssum[tid] = t;
    }
    __syncthreads();

    out[((size_t)b * NUM_USER + u) * 2 + o] = Ppow[0] * d1s[u * 2 + o] / ssum[o];
}

// ---------------------------------------------------------------------------
extern "C" void kernel_launch(void* const* d_in, const int* in_sizes, int n_in,
                              void* d_out, int out_size, void* d_ws, size_t ws_size,
                              hipStream_t stream)
{
    const float* D    = (const float*)d_in[0];
    const float* B    = (const float*)d_in[1];
    const float* Ppow = (const float*)d_in[2];
    // d_in[3] = D_0 (unused by the reference math)
    const float* W1 = (const float*)d_in[4];
    const float* b1 = (const float*)d_in[5];
    const float* W2 = (const float*)d_in[6];
    const float* b2 = (const float*)d_in[7];
    const float* W3 = (const float*)d_in[8];
    const float* b3 = (const float*)d_in[9];
    const float* W4 = (const float*)d_in[10];
    const float* b4 = (const float*)d_in[11];

    char* ws = (char*)d_ws;
    _Float16* X   = (_Float16*)(ws);                          //  2 MB
    _Float16* H1  = (_Float16*)(ws + ((size_t)2  << 20));     // 16 MB
    _Float16* H2  = (_Float16*)(ws + ((size_t)18 << 20));     // 16 MB
    _Float16* H3  = (_Float16*)(ws + ((size_t)34 << 20));     // 16 MB
    _Float16* W1t = (_Float16*)(ws + ((size_t)50 << 20));     // 64 KB
    _Float16* W2t = (_Float16*)(ws + ((size_t)51 << 20));     // 512 KB
    _Float16* W3t = (_Float16*)(ws + ((size_t)52 << 20));     // 512 KB

    // 1) fuse antenna-reduction + concat -> X f16 (bandwidth-bound pass over B)
    build_x_kernel<<<BATCH, 256, 0, stream>>>(D, B, X);

    // 2) weight convert+transpose (tiny, L2-resident thereafter)
    convert_wt_kernel<<<(K0 * NHID + 255) / 256, 256, 0, stream>>>(W1, W1t, K0, NHID);
    convert_wt_kernel<<<(NHID * NHID + 255) / 256, 256, 0, stream>>>(W2, W2t, NHID, NHID);
    convert_wt_kernel<<<(NHID * NHID + 255) / 256, 256, 0, stream>>>(W3, W3t, NHID, NHID);

    // 3) WMMA GEMM chain with fused bias+ReLU, async-DMA double-buffered tiles
    dim3 grid(NHID / TILE_N, MROWS / TILE_M);   // (4, 128)
    gemm_relu_wmma_kernel<<<grid, 256, 0, stream>>>(X,  W1t, b1, H1, K0,   NHID);
    gemm_relu_wmma_kernel<<<grid, 256, 0, stream>>>(H1, W2t, b2, H2, NHID, NHID);
    gemm_relu_wmma_kernel<<<grid, 256, 0, stream>>>(H2, W3t, b3, H3, NHID, NHID);

    // 4) 512->2 head + sigmoid + per-batch user normalization
    head_norm_kernel<<<BATCH, 64, 0, stream>>>(H3, W4, b4, Ppow, (float*)d_out);
}
